// ExpressionBert_56392920596769
// MI455X (gfx1250) — matrix operand order repaired and checked by reading, not compiled, not run on hardware
//
#include <hip/hip_runtime.h>
#include <hip/hip_bf16.h>
#include <math.h>

// ---------------- model dims ----------------
#define LAYERS 4
#define DMODEL 512
#define NHEAD  4
#define DHEAD  128
#define FFDIM  128
#define SEQ    256
#define NBATCH 128
#define MAXPOS 256
#define NFEATS 6
#define ROWS   (NBATCH * SEQ)   // 32768
#define BH     (NBATCH * NHEAD) // 512

typedef unsigned short ushort_t;
typedef __attribute__((ext_vector_type(16))) __bf16        bf16x16;
typedef __attribute__((ext_vector_type(8)))  float         f32x8;
typedef __attribute__((ext_vector_type(4)))  unsigned int  u32x4;

union Frag { u32x4 u[2]; bf16x16 v; };

__device__ __forceinline__ ushort_t f2bf(float f) {
  unsigned int u = __float_as_uint(f);
  u += 0x7FFFu + ((u >> 16) & 1u);   // round-to-nearest-even
  return (ushort_t)(u >> 16);
}

// ---------------------------------------------------------------------------
// Generic batched bf16 WMMA GEMM: C[b] = A[b] (MxK) * B[b] (KxN) (+bias[n])
// One wave computes a 32x32 tile as 2x2 WMMA sub-tiles (fragment reuse: each
// A/B fragment feeds two v_wmma_f32_16x16x32_bf16 -> 16 FLOP/byte fetched).
// Requires M % 32 == 0, N % 32 == 0, K % 32 == 0 (true for all call sites).
//
// A fragment (16x32, 16-bit): lane<16 -> M=lane, K = 0..7 & 16..23
//                             lane>=16 -> M=lane-16, K = 8..15 & 24..31
// B fragment (32x16, 16-bit): lane L -> col N=L&15, same K split as A; VGPR packs rows K,K+1
// C layout: VGPR r -> M = r + 8*(lane>>4), N = lane&15
// ---------------------------------------------------------------------------
__global__ __launch_bounds__(128) void wmma_gemm(
    const ushort_t* __restrict__ A, long lda, long aBatch,
    const ushort_t* __restrict__ Bm, long ldb, long bBatch,
    float* __restrict__ C, long cRow, long cCol, long cBatch,
    const float* __restrict__ bias,
    int M, int N, int K, int accum)
{
  const int lane = threadIdx.x & 31;
  const int wave = (int)(blockIdx.x * (blockDim.x >> 5)) + (int)(threadIdx.x >> 5);
  const int tilesN = N >> 5;          // 32-wide tiles
  const int tilesM = M >> 5;
  if (wave >= tilesM * tilesN) return;
  const int tm = wave / tilesN;
  const int tn = wave - tm * tilesN;
  const long batch = blockIdx.y;

  const ushort_t* Ab = A + batch * aBatch;
  const ushort_t* Bb = Bm + batch * bBatch;
  float*          Cb = C + batch * cBatch;

  const int m  = tm * 32 + (lane & 15);   // rows m and m+16
  const int kA = (lane >> 4) * 8;
  const int n  = tn * 32;                 // cols n..n+15 and n+16..n+31
  const ushort_t* ap0 = Ab + (long)m * lda + kA;
  const ushort_t* ap1 = ap0 + 16 * lda;
  const ushort_t* bp  = Bb + (long)lane * ldb + n;

  f32x8 acc00 = {}, acc01 = {}, acc10 = {}, acc11 = {};
#pragma unroll 2
  for (int k0 = 0; k0 < K; k0 += 32) {
    Frag fa0, fa1, fb0, fb1;
    fa0.u[0] = *(const u32x4*)(ap0 + k0);
    fa0.u[1] = *(const u32x4*)(ap0 + k0 + 16);
    fa1.u[0] = *(const u32x4*)(ap1 + k0);
    fa1.u[1] = *(const u32x4*)(ap1 + k0 + 16);
    const ushort_t* br = Bb + (long)(k0 + kA) * ldb + n + (lane & 15);
#pragma unroll
    for (int j = 0; j < 2; ++j)
#pragma unroll
      for (int i = 0; i < 4; ++i) {
        const ushort_t* p = br + (long)(16 * j + 2 * i) * ldb;
        fb0.u[j][i] = (unsigned)p[0] | ((unsigned)p[ldb] << 16);
        fb1.u[j][i] = (unsigned)p[16] | ((unsigned)p[ldb + 16] << 16);
      }
    __builtin_prefetch(ap0 + k0 + 32, 0, 1);   // global_prefetch_b8
    acc00 = __builtin_amdgcn_wmma_f32_16x16x32_bf16(
        false, fa0.v, false, fb0.v, (short)0, acc00, false, false);
    acc01 = __builtin_amdgcn_wmma_f32_16x16x32_bf16(
        false, fa0.v, false, fb1.v, (short)0, acc01, false, false);
    acc10 = __builtin_amdgcn_wmma_f32_16x16x32_bf16(
        false, fa1.v, false, fb0.v, (short)0, acc10, false, false);
    acc11 = __builtin_amdgcn_wmma_f32_16x16x32_bf16(
        false, fa1.v, false, fb1.v, (short)0, acc11, false, false);
  }

  const int hi  = lane >> 4;
  const int cn0 = n + (lane & 15);
  const int cn1 = cn0 + 16;
  const float bv0 = bias ? bias[cn0] : 0.0f;
  const float bv1 = bias ? bias[cn1] : 0.0f;
#pragma unroll
  for (int r = 0; r < 8; ++r) {
    const int cm0 = tm * 32 + r + 8 * hi;
    const int cm1 = cm0 + 16;
    const long o00 = (long)cm0 * cRow + (long)cn0 * cCol;
    const long o01 = (long)cm0 * cRow + (long)cn1 * cCol;
    const long o10 = (long)cm1 * cRow + (long)cn0 * cCol;
    const long o11 = (long)cm1 * cRow + (long)cn1 * cCol;
    if (accum) {
      Cb[o00] += acc00[r] + bv0;
      Cb[o01] += acc01[r] + bv1;
      Cb[o10] += acc10[r] + bv0;
      Cb[o11] += acc11[r] + bv1;
    } else {
      Cb[o00] = acc00[r] + bv0;
      Cb[o01] = acc01[r] + bv1;
      Cb[o10] = acc10[r] + bv0;
      Cb[o11] = acc11[r] + bv1;
    }
  }
}

// ---------------------------------------------------------------------------
// Embedding: x = LN(ids @ in_W + in_b + tok_emb); write fp32 x and bf16 copy.
// One 256-thread block per row (D=512 -> 2 columns/thread).
// ---------------------------------------------------------------------------
__global__ __launch_bounds__(256) void embed_kernel(
    const int* __restrict__ ids, const float* __restrict__ inW,
    const float* __restrict__ inb, const float* __restrict__ tok,
    const float* __restrict__ g, const float* __restrict__ bta,
    float* __restrict__ x, ushort_t* __restrict__ xb)
{
  __shared__ float red[256];
  __shared__ float red2[256];
  const long row = blockIdx.x;
  const int  t   = threadIdx.x;
  const int* idr = ids + row * NFEATS;
  float f[NFEATS];
#pragma unroll
  for (int i = 0; i < NFEATS; ++i) f[i] = (float)idr[i];

  float v[2];
#pragma unroll
  for (int p = 0; p < 2; ++p) {
    const int j = t + p * 256;
    float val = inb[j] + tok[j];
#pragma unroll
    for (int i = 0; i < NFEATS; ++i) val += f[i] * inW[i * DMODEL + j];
    v[p] = val;
  }
  red[t] = v[0] + v[1];
  red2[t] = v[0] * v[0] + v[1] * v[1];
  __syncthreads();
  for (int o = 128; o > 0; o >>= 1) {
    if (t < o) { red[t] += red[t + o]; red2[t] += red2[t + o]; }
    __syncthreads();
  }
  const float mu  = red[0] * (1.0f / DMODEL);
  const float var = red2[0] * (1.0f / DMODEL) - mu * mu;
  const float rs  = rsqrtf(var + 1e-12f);
#pragma unroll
  for (int p = 0; p < 2; ++p) {
    const int j = t + p * 256;
    const float val = (v[p] - mu) * rs * g[j] + bta[j];
    x[row * DMODEL + j]  = val;
    xb[row * DMODEL + j] = f2bf(val);
  }
}

// ---------------------------------------------------------------------------
// Residual + LayerNorm: x = LN(x + y); y already includes GEMM bias.
// ---------------------------------------------------------------------------
__global__ __launch_bounds__(256) void add_ln_kernel(
    float* __restrict__ x, const float* __restrict__ y,
    const float* __restrict__ g, const float* __restrict__ bta,
    ushort_t* __restrict__ xb)
{
  __shared__ float red[256];
  __shared__ float red2[256];
  const long row = blockIdx.x;
  const int  t   = threadIdx.x;
  float v[2];
#pragma unroll
  for (int p = 0; p < 2; ++p) {
    const int j = t + p * 256;
    v[p] = x[row * DMODEL + j] + y[row * DMODEL + j];
  }
  red[t] = v[0] + v[1];
  red2[t] = v[0] * v[0] + v[1] * v[1];
  __syncthreads();
  for (int o = 128; o > 0; o >>= 1) {
    if (t < o) { red[t] += red[t + o]; red2[t] += red2[t + o]; }
    __syncthreads();
  }
  const float mu  = red[0] * (1.0f / DMODEL);
  const float var = red2[0] * (1.0f / DMODEL) - mu * mu;
  const float rs  = rsqrtf(var + 1e-12f);
#pragma unroll
  for (int p = 0; p < 2; ++p) {
    const int j = t + p * 256;
    const float val = (v[p] - mu) * rs * g[j] + bta[j];
    x[row * DMODEL + j]  = val;
    xb[row * DMODEL + j] = f2bf(val);
  }
}

// ---------------------------------------------------------------------------
// Scatter a [B,S,H,DH] fp32 projection into bf16 [BH,S,DH] (+ optional
// transposed [BH,DH,S] copy used as the B operand of q@k^T).
// ---------------------------------------------------------------------------
__global__ void scatter_heads(const float* __restrict__ src,
                              ushort_t* __restrict__ dst,
                              ushort_t* __restrict__ dstT)
{
  const long idx = (long)blockIdx.x * blockDim.x + threadIdx.x;
  if (idx >= (long)ROWS * DMODEL) return;
  const int  d  = (int)(idx % DHEAD);
  const long r1 = idx / DHEAD;
  const int  h  = (int)(r1 % NHEAD);
  const long r2 = r1 / NHEAD;
  const int  s  = (int)(r2 % SEQ);
  const int  b  = (int)(r2 / SEQ);
  const ushort_t bf = f2bf(src[idx]);
  const long bh = (long)b * NHEAD + h;
  dst[(bh * SEQ + s) * DHEAD + d] = bf;
  if (dstT) dstT[(bh * DHEAD + d) * SEQ + s] = bf;
}

// ---------------------------------------------------------------------------
// pe gather: pe[l][r][d] = dist_emb[l-r+MAXPOS-1][d], written in two
// transposed bf16 layouts: peA[l][d][r] (for q.pe) and peB[r][d][l] (for k.pe)
// ---------------------------------------------------------------------------
__global__ void pe_gather(const float* __restrict__ dist,
                          ushort_t* __restrict__ peA,
                          ushort_t* __restrict__ peB)
{
  const long idx = (long)blockIdx.x * blockDim.x + threadIdx.x;
  const long total = (long)SEQ * SEQ * DHEAD;
  if (idx >= total) return;
  const int  d  = (int)(idx % DHEAD);
  const long r1 = idx / DHEAD;
  const int  r  = (int)(r1 % SEQ);
  const int  l  = (int)(r1 / SEQ);
  const float val = dist[(long)(l - r + MAXPOS - 1) * DHEAD + d];
  const ushort_t bf = f2bf(val);
  peA[((long)l * DHEAD + d) * SEQ + r] = bf;
  peB[((long)r * DHEAD + d) * SEQ + l] = bf;
}

// ---------------------------------------------------------------------------
// Softmax over keys: one 256-thread block per (bh, l) row, S == blockDim.
// ---------------------------------------------------------------------------
__global__ __launch_bounds__(256) void softmax_kernel(
    const float* __restrict__ scores, const float* __restrict__ mask,
    ushort_t* __restrict__ probs, float scale)
{
  __shared__ float red[256];
  const long row = blockIdx.x;          // bh*SEQ + l
  const int  t   = threadIdx.x;         // key index r
  const long bh  = row / SEQ;
  const int  b   = (int)(bh / NHEAD);
  float s = scores[row * SEQ + t] * scale +
            (1.0f - mask[(long)b * SEQ + t]) * (-1.0e9f);
  red[t] = s; __syncthreads();
  for (int o = 128; o > 0; o >>= 1) {
    if (t < o) red[t] = fmaxf(red[t], red[t + o]);
    __syncthreads();
  }
  const float mx = red[0]; __syncthreads();
  const float e = __expf(s - mx);
  red[t] = e; __syncthreads();
  for (int o = 128; o > 0; o >>= 1) {
    if (t < o) red[t] += red[t + o];
    __syncthreads();
  }
  probs[row * SEQ + t] = f2bf(e / red[0]);
}

// ctx fp32 [bh][l][d] -> bf16 rows [b*S+l][h*DH+d] for the O projection
__global__ void ctx_to_rows(const float* __restrict__ src,
                            ushort_t* __restrict__ dst)
{
  const long idx = (long)blockIdx.x * blockDim.x + threadIdx.x;
  if (idx >= (long)BH * SEQ * DHEAD) return;
  const int  d  = (int)(idx % DHEAD);
  const long r1 = idx / DHEAD;
  const int  l  = (int)(r1 % SEQ);
  const long bh = r1 / SEQ;
  const int  b  = (int)(bh / NHEAD);
  const int  h  = (int)(bh % NHEAD);
  dst[((long)b * SEQ + l) * DMODEL + h * DHEAD + d] = f2bf(src[idx]);
}

// exact GELU -> bf16
__global__ void gelu_kernel(const float* __restrict__ src,
                            ushort_t* __restrict__ dst, long n)
{
  const long idx = (long)blockIdx.x * blockDim.x + threadIdx.x;
  if (idx >= n) return;
  const float v = src[idx];
  dst[idx] = f2bf(0.5f * v * (1.0f + erff(v * 0.70710678118654752f)));
}

__global__ void f32_to_bf16_kernel(const float* __restrict__ src,
                                   ushort_t* __restrict__ dst, long n)
{
  const long idx = (long)blockIdx.x * blockDim.x + threadIdx.x;
  if (idx < n) dst[idx] = f2bf(src[idx]);
}

// ---------------------------------------------------------------------------
extern "C" void kernel_launch(void* const* d_in, const int* in_sizes, int n_in,
                              void* d_out, int out_size, void* d_ws, size_t ws_size,
                              hipStream_t stream)
{
  (void)in_sizes; (void)n_in; (void)out_size; (void)ws_size;
  const int*   input_ids = (const int*)  d_in[0];
  const float* attn_mask = (const float*)d_in[1];
  const float* in_W  = (const float*)d_in[2];
  const float* in_b  = (const float*)d_in[3];
  const float* tok   = (const float*)d_in[4];
  const float* eg    = (const float*)d_in[5];
  const float* ebb   = (const float*)d_in[6];
  const float* dist  = (const float*)d_in[7];
  const float* Wq = (const float*)d_in[8];  const float* bq = (const float*)d_in[9];
  const float* Wk = (const float*)d_in[10]; const float* bk = (const float*)d_in[11];
  const float* Wv = (const float*)d_in[12]; const float* bv = (const float*)d_in[13];
  const float* Wo = (const float*)d_in[14]; const float* bo = (const float*)d_in[15];
  const float* g1 = (const float*)d_in[16]; const float* b1 = (const float*)d_in[17];
  const float* Wi = (const float*)d_in[18]; const float* bi = (const float*)d_in[19];
  const float* Wo2= (const float*)d_in[20]; const float* bo2= (const float*)d_in[21];
  const float* g2 = (const float*)d_in[22]; const float* b2 = (const float*)d_in[23];

  float* x = (float*)d_out;   // fp32 activations live in d_out (it IS the output)

  // ---- workspace bump allocator ----
  char* wp = (char*)d_ws;
  auto alloc = [&](size_t bytes) -> void* {
    void* p = (void*)wp; wp += (bytes + 255) & ~(size_t)255; return p;
  };
  ushort_t* wqb  = (ushort_t*)alloc((size_t)LAYERS * DMODEL * DMODEL * 2);
  ushort_t* wkb  = (ushort_t*)alloc((size_t)LAYERS * DMODEL * DMODEL * 2);
  ushort_t* wvb  = (ushort_t*)alloc((size_t)LAYERS * DMODEL * DMODEL * 2);
  ushort_t* wob  = (ushort_t*)alloc((size_t)LAYERS * DMODEL * DMODEL * 2);
  ushort_t* wib  = (ushort_t*)alloc((size_t)LAYERS * DMODEL * FFDIM  * 2);
  ushort_t* wo2b = (ushort_t*)alloc((size_t)LAYERS * FFDIM  * DMODEL * 2);
  ushort_t* peA  = (ushort_t*)alloc((size_t)SEQ * DHEAD * SEQ * 2);
  ushort_t* peB  = (ushort_t*)alloc((size_t)SEQ * DHEAD * SEQ * 2);
  ushort_t* xb   = (ushort_t*)alloc((size_t)ROWS * DMODEL * 2);
  ushort_t* qb   = (ushort_t*)alloc((size_t)BH * SEQ * DHEAD * 2);
  ushort_t* kb   = (ushort_t*)alloc((size_t)BH * SEQ * DHEAD * 2);
  ushort_t* kt   = (ushort_t*)alloc((size_t)BH * DHEAD * SEQ * 2);
  ushort_t* vb   = (ushort_t*)alloc((size_t)BH * SEQ * DHEAD * 2);
  float*    scores = (float*)alloc((size_t)BH * SEQ * SEQ * 4);
  ushort_t* probsb = (ushort_t*)alloc((size_t)BH * SEQ * SEQ * 2);
  float*    gbuf   = (float*)alloc((size_t)ROWS * DMODEL * 4);  // reused scratch
  ushort_t* ctxb   = (ushort_t*)alloc((size_t)ROWS * DMODEL * 2);
  ushort_t* hb     = (ushort_t*)alloc((size_t)ROWS * FFDIM * 2);

  auto conv = [&](const float* s, ushort_t* d, long n) {
    f32_to_bf16_kernel<<<dim3((unsigned)((n + 255) / 256)), 256, 0, stream>>>(s, d, n);
  };
  auto gemm = [&](const ushort_t* A, long lda, long aB,
                  const ushort_t* Bm, long ldb, long bB,
                  float* C, long cR, long cC, long cB,
                  const float* bias, int M, int N, int K, int accum, int batches) {
    const int waves = (M / 32) * (N / 32);          // 32x32 tile per wave
    const int blocks = (waves + 3) / 4;             // 4 waves (128 thr) / block
    wmma_gemm<<<dim3((unsigned)blocks, (unsigned)batches), 128, 0, stream>>>(
        A, lda, aB, Bm, ldb, bB, C, cR, cC, cB, bias, M, N, K, accum);
  };

  // ---- one-time (per launch) weight / pe conversion ----
  conv(Wq,  wqb,  (long)LAYERS * DMODEL * DMODEL);
  conv(Wk,  wkb,  (long)LAYERS * DMODEL * DMODEL);
  conv(Wv,  wvb,  (long)LAYERS * DMODEL * DMODEL);
  conv(Wo,  wob,  (long)LAYERS * DMODEL * DMODEL);
  conv(Wi,  wib,  (long)LAYERS * DMODEL * FFDIM);
  conv(Wo2, wo2b, (long)LAYERS * FFDIM * DMODEL);
  {
    const long n = (long)SEQ * SEQ * DHEAD;
    pe_gather<<<dim3((unsigned)((n + 255) / 256)), 256, 0, stream>>>(dist, peA, peB);
  }

  // ---- embedding + LN ----
  embed_kernel<<<dim3(ROWS), 256, 0, stream>>>(input_ids, in_W, in_b, tok, eg, ebb, x, xb);

  const float scale = 0.08838834764831845f;  // 1/sqrt(128)
  const long nProj = (long)ROWS * DMODEL;
  const unsigned gProj = (unsigned)((nProj + 255) / 256);

  for (int l = 0; l < LAYERS; ++l) {
    const ushort_t* wq_l  = wqb  + (size_t)l * DMODEL * DMODEL;
    const ushort_t* wk_l  = wkb  + (size_t)l * DMODEL * DMODEL;
    const ushort_t* wv_l  = wvb  + (size_t)l * DMODEL * DMODEL;
    const ushort_t* wo_l  = wob  + (size_t)l * DMODEL * DMODEL;
    const ushort_t* wi_l  = wib  + (size_t)l * DMODEL * FFDIM;
    const ushort_t* wo2_l = wo2b + (size_t)l * FFDIM * DMODEL;

    // Q / K / V projections (bias fused); scatter into head layouts (bf16)
    gemm(xb, DMODEL, 0, wq_l, DMODEL, 0, gbuf, DMODEL, 1, 0,
         bq + (size_t)l * DMODEL, ROWS, DMODEL, DMODEL, 0, 1);
    scatter_heads<<<gProj, 256, 0, stream>>>(gbuf, qb, nullptr);

    gemm(xb, DMODEL, 0, wk_l, DMODEL, 0, gbuf, DMODEL, 1, 0,
         bk + (size_t)l * DMODEL, ROWS, DMODEL, DMODEL, 0, 1);
    scatter_heads<<<gProj, 256, 0, stream>>>(gbuf, kb, kt);

    gemm(xb, DMODEL, 0, wv_l, DMODEL, 0, gbuf, DMODEL, 1, 0,
         bv + (size_t)l * DMODEL, ROWS, DMODEL, DMODEL, 0, 1);
    scatter_heads<<<gProj, 256, 0, stream>>>(gbuf, vb, nullptr);

    // scores = q @ k^T                (batched over bh)
    gemm(qb, DHEAD, (long)SEQ * DHEAD,
         kt, SEQ,   (long)DHEAD * SEQ,
         scores, SEQ, 1, (long)SEQ * SEQ,
         nullptr, SEQ, SEQ, DHEAD, 0, BH);
    // scores += q . pe               (batched over query index l)
    gemm(qb, (long)SEQ * DHEAD, DHEAD,
         peA, SEQ, (long)DHEAD * SEQ,
         scores, (long)SEQ * SEQ, 1, SEQ,
         nullptr, BH, SEQ, DHEAD, 1, SEQ);
    // scores += k . pe               (batched over key index r)
    gemm(kb, (long)SEQ * DHEAD, DHEAD,
         peB, SEQ, (long)DHEAD * SEQ,
         scores, (long)SEQ * SEQ, SEQ, 1,
         nullptr, BH, SEQ, DHEAD, 1, SEQ);

    // softmax -> bf16 probabilities
    softmax_kernel<<<dim3(BH * SEQ), 256, 0, stream>>>(scores, attn_mask, probsb, scale);

    // ctx = probs @ v                (batched over bh) -> fp32 [bh][l][d]
    gemm(probsb, SEQ, (long)SEQ * SEQ,
         vb, DHEAD, (long)SEQ * DHEAD,
         gbuf, DHEAD, 1, (long)SEQ * DHEAD,
         nullptr, SEQ, DHEAD, SEQ, 0, BH);
    {
      const long n = (long)BH * SEQ * DHEAD;
      ctx_to_rows<<<dim3((unsigned)((n + 255) / 256)), 256, 0, stream>>>(gbuf, ctxb);
    }

    // attention output projection + residual + LN
    gemm(ctxb, DMODEL, 0, wo_l, DMODEL, 0, gbuf, DMODEL, 1, 0,
         bo + (size_t)l * DMODEL, ROWS, DMODEL, DMODEL, 0, 1);
    add_ln_kernel<<<dim3(ROWS), 256, 0, stream>>>(
        x, gbuf, g1 + (size_t)l * DMODEL, b1 + (size_t)l * DMODEL, xb);

    // FFN: gelu(x @ Wi + bi) @ Wo2 + bo2, residual + LN
    gemm(xb, DMODEL, 0, wi_l, FFDIM, 0, gbuf, FFDIM, 1, 0,
         bi + (size_t)l * FFDIM, ROWS, FFDIM, DMODEL, 0, 1);
    {
      const long n = (long)ROWS * FFDIM;
      gelu_kernel<<<dim3((unsigned)((n + 255) / 256)), 256, 0, stream>>>(gbuf, hb, n);
    }
    gemm(hb, FFDIM, 0, wo2_l, DMODEL, 0, gbuf, DMODEL, 1, 0,
         bo2 + (size_t)l * DMODEL, ROWS, DMODEL, FFDIM, 0, 1);
    add_ln_kernel<<<dim3(ROWS), 256, 0, stream>>>(
        x, gbuf, g2 + (size_t)l * DMODEL, b2 + (size_t)l * DMODEL, xb);
  }
}
